// GraphSage_62783831933363
// MI455X (gfx1250) — matrix operand order repaired and checked by reading, not compiled, not run on hardware
//
#include <hip/hip_runtime.h>
#include <hip/hip_bf16.h>
#include <math.h>

// ---------------------------------------------------------------------------
// GraphSAGE (3x SAGEConv project+mean+normalize, 3x FC) for MI455X (gfx1250).
//  * All GEMMs run on v_wmma_f32_16x16x32_bf16 (wave32), fp32 accumulate.
//  * segment_mean(xp)@Wl == D^-1 * segment_sum(xp@Wl): GEMM before scatter,
//    aggregation in 256-dim space (10x less gather traffic than 2613-dim).
//  * Block tile 128M x 128N, 8 waves; each wave = 16 rows x 8 WMMA n-tiles.
//    All 8 B fragments are loaded into distinct registers before the WMMA
//    sequence so ds_loads pipeline instead of serializing per-WMMA.
//  * B tile (8KB/k-step) shared by all waves -> staged in LDS (double-
//    buffered, one barrier per k-step): 8x less redundant L2->WGP traffic.
//  * A fragments load straight from global (unique per wave, L2-resident;
//    A bf16 = 54MB, W bf16 <= 14.5MB, both fit the 192MB L2).
//  * Weights pre-packed to WMMA B-fragment order (per-lane contiguous 32B).
//  * K padded to 2688 (=84*32), N padded to mult of 128 -> no tails/guards.
// ---------------------------------------------------------------------------

typedef unsigned short u16;
typedef __attribute__((ext_vector_type(16))) __bf16 v16bf;
typedef __attribute__((ext_vector_type(8)))  __bf16 v8bf;
typedef __attribute__((ext_vector_type(8)))  float  v8f;

enum { MODE_F32 = 0, MODE_RELU_BF16 = 1, MODE_ELU_BF16 = 2 };

__device__ __forceinline__ u16 f2bf(float f) {
  unsigned u = __float_as_uint(f);
  u += 0x7FFFu + ((u >> 16) & 1u);   // round-to-nearest-even
  return (u16)(u >> 16);
}

// --- fp32 -> bf16 with column zero-padding (row stride K -> Kpad) ----------
__global__ void convert_pad_bf16(const float* __restrict__ X, int K, int Kpad,
                                 u16* __restrict__ out, long total) {
  long idx = (long)blockIdx.x * blockDim.x + threadIdx.x;
  if (idx >= total) return;
  long row = idx / Kpad;
  int  col = (int)(idx - row * Kpad);
  float v = (col < K) ? X[row * K + col] : 0.0f;
  out[idx] = f2bf(v);
}

// --- pack fp32 weight (K x Nn, row-major) into WMMA B-fragment tiles -------
// Tile (kt, nt) covers B[kt*32 .. +31][nt*16 .. +15]; 512 bf16 per tile,
// laid out so lane l reads 16 contiguous bf16 (=32B, two b128 loads):
//   element i of lane l = B[kt*32 + (l<16?0:16) + i][nt*16 + (l&15)]
__global__ void pack_w_bf16(const float* __restrict__ W, int K, int Nn,
                            int ktiles, u16* __restrict__ out, long total) {
  long idx = (long)blockIdx.x * blockDim.x + threadIdx.x;
  if (idx >= total) return;
  int tile = (int)(idx >> 9);
  int r    = (int)(idx & 511);
  int l = r >> 4, i = r & 15;
  int nt = tile / ktiles;
  int kt = tile - nt * ktiles;
  int k = kt * 32 + ((l & 16) ? 16 : 0) + i;
  int n = nt * 16 + (l & 15);
  float v = (k < K && n < Nn) ? W[(size_t)k * Nn + n] : 0.0f;
  out[idx] = f2bf(v);
}

// --- bf16 WMMA GEMM: C[M x Npad] = act(A[M x Kpad] @ Bpacked + bias) -------
// 8 waves / block cover 128M x 128N; wave w -> rows m0 = by*128 + w*16 and
// eight 16x16 n-tiles. B tile (8KB/k-step) staged in LDS, double-buffered.
__global__ __launch_bounds__(256) void gemm_bf16_wmma(
    const u16* __restrict__ A, int lda,
    const u16* __restrict__ Bp, int ktiles,
    const float* __restrict__ bias, int nbias,
    void* __restrict__ Out, int ldo,
    int M, int mode)
{
  __shared__ __align__(32) u16 sB[2][8 * 512];   // 2 x 8KB

  const int tid  = threadIdx.x;
  const int lane = tid & 31;
  const int w    = tid >> 5;
  int m0 = blockIdx.y * 128 + w * 16;
  const bool active = (m0 < M);                  // M is a multiple of 16
  if (!active) m0 = 0;                           // clamp; still join barriers
  const int n0  = blockIdx.x * 128;
  const int nt0 = n0 >> 4;

  // A fragment addressing (contiguous k order, not the ISA table order):
  // lanes 0-15: K 0-15 ; lanes 16-31: K 16-31 (matches packed B)
  const u16* arow = A + (size_t)(m0 + (lane & 15)) * lda;
  const int  khi  = (lane & 16) ? 16 : 0;

  // Cooperative B loader: thread t covers n-tile cc = t/32, 16 u16 at rr.
  const int cc = tid >> 5;               // 0..7
  const int rr = (tid & 31) * 16;        // 0..496
  const u16* gB = Bp + ((size_t)(nt0 + cc) * ktiles) * 512 + rr;

  // preload kt = 0
  *(v16bf*)(&sB[0][cc * 512 + rr]) = *(const v16bf*)(gB);

  v8f acc[8] = {};
  v16bf breg;
  for (int kt = 0; kt < ktiles; ++kt) {
    __syncthreads();                 // sB[kt&1] ready; safe to fill the other
    const bool more = (kt + 1 < ktiles);
    if (more) breg = *(const v16bf*)(gB + (size_t)(kt + 1) * 512);

    const int kb = kt * 32 + khi;
    v8bf alo = *(const v8bf*)(arow + kb);
    v8bf ahi = *(const v8bf*)(arow + kb + 8);
    if (more) __builtin_prefetch(arow + kb + 32, 0, 3);
    v16bf a = __builtin_shufflevector(alo, ahi,
              0,1,2,3,4,5,6,7,8,9,10,11,12,13,14,15);

    // Load all 8 B fragments first (distinct registers) so the 16
    // ds_load_b128 pipeline; then issue the 8 WMMAs back-to-back.
    const u16* sb = &sB[kt & 1][lane * 16];
    v16bf b[8];
#pragma unroll
    for (int c = 0; c < 8; ++c) b[c] = *(const v16bf*)(sb + c * 512);
#pragma unroll
    for (int c = 0; c < 8; ++c)
      acc[c] = __builtin_amdgcn_wmma_f32_16x16x32_bf16(
          false, a, false, b[c], (short)0, acc[c], false, false);

    if (more) *(v16bf*)(&sB[(kt + 1) & 1][cc * 512 + rr]) = breg;
  }

  if (!active) return;

  // C/D layout: lane l, VGPR j -> M = m0 + j + (l<16?0:8), N = n tile + (l&15)
  const int mbase = m0 + ((lane & 16) ? 8 : 0);
  const int nl = lane & 15;
#pragma unroll
  for (int c = 0; c < 8; ++c) {
    const int n = n0 + c * 16 + nl;
    const float bv = (bias != nullptr && n < nbias) ? bias[n] : 0.0f;
#pragma unroll
    for (int j = 0; j < 8; ++j) {
      if (n < ldo) {
        float v = acc[c][j] + bv;
        const size_t o = (size_t)(mbase + j) * ldo + n;
        if (mode == MODE_F32) {
          ((float*)Out)[o] = v;
        } else {
          if (mode == MODE_RELU_BF16) v = fmaxf(v, 0.0f);
          else                        v = (v > 0.0f) ? v : (expf(v) - 1.0f);
          ((u16*)Out)[o] = f2bf(v);
        }
      }
    }
  }
}

// --- degree: deg[dst] += 1 --------------------------------------------------
__global__ void scatter_deg(const int* __restrict__ ei, int E,
                            float* __restrict__ deg) {
  int e = blockIdx.x * blockDim.x + threadIdx.x;
  if (e < E) atomicAdd(&deg[ei[E + e]], 1.0f);
}

// --- acc[dst] += yl[src]  (256 floats / edge, 4 per thread) ----------------
__global__ void scatter_add(const int* __restrict__ ei, int E,
                            const float* __restrict__ yl,
                            float* __restrict__ acc) {
  long t = (long)blockIdx.x * blockDim.x + threadIdx.x;
  if (t >= (long)E * 64) return;
  int e = (int)(t >> 6), q = (int)(t & 63);
  int s = ei[e], d = ei[E + e];
  const float4 v = *(const float4*)(yl + (size_t)s * 256 + q * 4);
  float* a = acc + (size_t)d * 256 + q * 4;
  atomicAdd(a + 0, v.x); atomicAdd(a + 1, v.y);
  atomicAdd(a + 2, v.z); atomicAdd(a + 3, v.w);
}

// --- out = elu(l2norm(acc/max(deg,1) + bl + yr)) -> bf16 (stride 256) ------
__global__ __launch_bounds__(256) void combine_norm_elu(
    const float* __restrict__ acc, const float* __restrict__ deg,
    const float* __restrict__ yr, const float* __restrict__ bl,
    u16* __restrict__ hout) {
  const int v = blockIdx.x, f = threadIdx.x;
  const size_t off = (size_t)v * 256 + f;
  float t = acc[off] / fmaxf(deg[v], 1.0f) + bl[f] + yr[off];
  __shared__ float red[256];
  red[f] = t * t;
  __syncthreads();
  for (int s = 128; s > 0; s >>= 1) {
    if (f < s) red[f] += red[f + s];
    __syncthreads();
  }
  float x = t / fmaxf(sqrtf(red[0]), 1e-12f);
  float h = (x > 0.0f) ? x : (expf(x) - 1.0f);
  hout[off] = f2bf(h);
}

// ---------------------------------------------------------------------------
extern "C" void kernel_launch(void* const* d_in, const int* in_sizes, int n_in,
                              void* d_out, int out_size, void* d_ws, size_t ws_size,
                              hipStream_t stream) {
  constexpr int N = 10000, E = 160000, FIN = 2613;
  constexpr int KP1 = 2688;                  // 84*32 = 21*128: K & N pad for dim 2613
  constexpr int H = 256, OUTD = 100, NPF = 128;

  const float* x  = (const float*)d_in[0];
  const int*   ei = (const int*)d_in[1];
  const float* Wp[3] = {(const float*)d_in[2],  (const float*)d_in[7],  (const float*)d_in[12]};
  const float* bp[3] = {(const float*)d_in[3],  (const float*)d_in[8],  (const float*)d_in[13]};
  const float* Wl[3] = {(const float*)d_in[4],  (const float*)d_in[9],  (const float*)d_in[14]};
  const float* bl[3] = {(const float*)d_in[5],  (const float*)d_in[10], (const float*)d_in[15]};
  const float* Wr[3] = {(const float*)d_in[6],  (const float*)d_in[11], (const float*)d_in[16]};
  const float* Wf1 = (const float*)d_in[17]; const float* bf1 = (const float*)d_in[18];
  const float* Wf2 = (const float*)d_in[19]; const float* bf2 = (const float*)d_in[20];
  const float* Wf3 = (const float*)d_in[21]; const float* bf3 = (const float*)d_in[22];

  // Workspace layout (~155 MB total)
  char* base = (char*)d_ws; size_t off = 0;
  auto alloc = [&](size_t bytes) -> void* {
    void* p = base + off; off = (off + bytes + 255) & ~(size_t)255; return p;
  };
  u16*   xb  = (u16*)  alloc((size_t)N * KP1 * 2);  // activations (bf16, padded)
  u16*   xpb = (u16*)  alloc((size_t)N * KP1 * 2);  // projected feats (bf16)
  u16*   wpk = (u16*)  alloc((size_t)KP1 * KP1 * 2);// packed weight staging
  float* yl  = (float*)alloc((size_t)N * H * 4);
  float* yr  = (float*)alloc((size_t)N * H * 4);
  float* acc = (float*)alloc((size_t)N * H * 4);
  float* deg = (float*)alloc((size_t)N * 4);

  auto cdiv = [](long a, long b) { return (int)((a + b - 1) / b); };

  { // x (fp32, N x 2613) -> xb (bf16, N x 2688, zero-padded)
    long tot = (long)N * KP1;
    convert_pad_bf16<<<cdiv(tot, 256), 256, 0, stream>>>(x, FIN, KP1, xb, tot);
  }

  auto pack = [&](const float* W, int K, int Nn, int Kpad, int Npad) {
    long tot = (long)Kpad * Npad;
    pack_w_bf16<<<cdiv(tot, 256), 256, 0, stream>>>(W, K, Nn, Kpad / 32, wpk, tot);
  };

  const dim3 gH(H / 128, cdiv(N, 128));           // 2 x 79

  auto sage = [&](int li, int Kreal, int Kpad) {
    const int ktiles = Kpad / 32;
    // xp = relu(h @ Wp + bp)      (din -> din, bf16 out, padded stride)
    pack(Wp[li], Kreal, Kreal, Kpad, Kpad);
    gemm_bf16_wmma<<<dim3(Kpad / 128, cdiv(N, 128)), 256, 0, stream>>>(
        xb, Kpad, wpk, ktiles, bp[li], Kreal, xpb, Kpad, N, MODE_RELU_BF16);
    // yl = xp @ Wl (bias added after aggregation)
    pack(Wl[li], Kreal, H, Kpad, H);
    gemm_bf16_wmma<<<gH, 256, 0, stream>>>(
        xpb, Kpad, wpk, ktiles, nullptr, 0, yl, H, N, MODE_F32);
    // yr = xp @ Wr
    pack(Wr[li], Kreal, H, Kpad, H);
    gemm_bf16_wmma<<<gH, 256, 0, stream>>>(
        xpb, Kpad, wpk, ktiles, nullptr, 0, yr, H, N, MODE_F32);
    // mean aggregation in 256-dim output space
    (void)hipMemsetAsync(acc, 0, (size_t)N * H * 4, stream);
    (void)hipMemsetAsync(deg, 0, (size_t)N * 4, stream);
    scatter_deg<<<cdiv(E, 256), 256, 0, stream>>>(ei, E, deg);
    scatter_add<<<cdiv((long)E * 64, 256), 256, 0, stream>>>(ei, E, yl, acc);
    // h = elu(l2norm(acc/deg + bl + yr)) -> xb (bf16, stride 256)
    combine_norm_elu<<<N, 256, 0, stream>>>(acc, deg, yr, bl[li], xb);
  };

  sage(0, FIN, KP1);
  sage(1, H, H);
  sage(2, H, H);

  // FC head
  pack(Wf1, H, H, H, H);
  gemm_bf16_wmma<<<gH, 256, 0, stream>>>(xb, H, wpk, H / 32, bf1, H, xpb, H, N, MODE_ELU_BF16);
  pack(Wf2, H, H, H, H);
  gemm_bf16_wmma<<<gH, 256, 0, stream>>>(xpb, H, wpk, H / 32, bf2, H, xb, H, N, MODE_ELU_BF16);
  pack(Wf3, H, OUTD, H, NPF);
  gemm_bf16_wmma<<<dim3(NPF / 128, cdiv(N, 128)), 256, 0, stream>>>(
      xb, H, wpk, H / 32, bf3, OUTD, d_out, OUTD, N, MODE_F32);
}
